// GCN_26190710571463
// MI455X (gfx1250) — matrix operand order repaired and checked by reading, not compiled, not run on hardware
//
#include <hip/hip_runtime.h>

typedef __attribute__((ext_vector_type(8)))  _Float16 v8h;
typedef __attribute__((ext_vector_type(16))) _Float16 v16h;
typedef __attribute__((ext_vector_type(8)))  float    v8f;

#define EPSV 1e-5f

// ---------------- utility kernels ----------------

__global__ void zero_kernel(float* __restrict__ p, long n) {
    long i = (long)blockIdx.x * blockDim.x + threadIdx.x;
    if (i < n) p[i] = 0.0f;
}

__global__ void deg_kernel(const float* __restrict__ w, const int* __restrict__ dst,
                           float* __restrict__ deg, int E) {
    int i = blockIdx.x * blockDim.x + threadIdx.x;
    if (i < E) atomicAdd(&deg[dst[i]], w[i]);
}

__global__ void dis_kernel(const float* __restrict__ deg, float* __restrict__ dis, int n) {
    int i = blockIdx.x * blockDim.x + threadIdx.x;
    if (i >= n) return;
    float d = deg[i];
    dis[i] = (d > 0.0f) ? rsqrtf(fmaxf(d, EPSV)) : 0.0f;
}

__global__ void norm_kernel(const int* __restrict__ src, const int* __restrict__ dst,
                            const float* __restrict__ w, const float* __restrict__ dis,
                            float* __restrict__ nrm, int E) {
    int i = blockIdx.x * blockDim.x + threadIdx.x;
    if (i >= E) return;
    nrm[i] = dis[src[i]] * w[i] * dis[dst[i]];
}

// ---------------- batchnorm ----------------

__global__ void bn_stats_kernel(const float* __restrict__ x, float* __restrict__ sums,
                                int nrows, int D, int rowsPerBlock) {
    int tid = threadIdx.x;
    int f = tid % D;
    int rOff = tid / D;
    int rStep = blockDim.x / D;
    int r0 = blockIdx.x * rowsPerBlock;
    int r1 = r0 + rowsPerBlock; if (r1 > nrows) r1 = nrows;
    float s = 0.f, q = 0.f;
    for (int r = r0 + rOff; r < r1; r += rStep) {
        float v = x[(long)r * D + f];
        s += v; q += v * v;
    }
    atomicAdd(&sums[f], s);
    atomicAdd(&sums[D + f], q);
}

__global__ void bn_finalize_kernel(const float* __restrict__ sums, const float* __restrict__ gamma,
                                   const float* __restrict__ beta, float* __restrict__ scale,
                                   float* __restrict__ shift, int D, float invN) {
    int f = blockIdx.x * blockDim.x + threadIdx.x;
    if (f >= D) return;
    float mean = sums[f] * invN;
    float var  = sums[D + f] * invN - mean * mean;
    float rstd = rsqrtf(var + EPSV);
    float sc = rstd * gamma[f];
    scale[f] = sc;
    shift[f] = beta[f] - mean * sc;
}

__global__ void bn_apply_split_kernel(const float* __restrict__ x, const float* __restrict__ scale,
                                      const float* __restrict__ shift, _Float16* __restrict__ hh,
                                      _Float16* __restrict__ hl, long n, int D) {
    long i = (long)blockIdx.x * blockDim.x + threadIdx.x;
    if (i >= n) return;
    int f = (int)(i % D);
    float v = fmaf(x[i], scale[f], shift[f]);
    _Float16 hi = (_Float16)v;
    hh[i] = hi;
    hl[i] = (_Float16)(v - (float)hi);
}

// ---------------- weight prep: transpose [K x Nout] -> [Nout x K] + f16 split ----------------

__global__ void transpose_split_kernel(const float* __restrict__ W, _Float16* __restrict__ bh,
                                       _Float16* __restrict__ bl, int K, int Nout) {
    int i = blockIdx.x * blockDim.x + threadIdx.x;
    if (i >= K * Nout) return;
    int k = i / Nout, n = i % Nout;
    float v = W[i];
    _Float16 hi = (_Float16)v;
    bh[(long)n * K + k] = hi;
    bl[(long)n * K + k] = (_Float16)(v - (float)hi);
}

// ---------------- WMMA GEMM: C[M x NT*16] = (Ah+Al)[M x K] @ (Bh+Bl)^T, Bt is [Nout x K] ----------------

template<int NT>
__global__ __launch_bounds__(256) void gemm_wmma_split(
    const _Float16* __restrict__ Ah, const _Float16* __restrict__ Al,
    const _Float16* __restrict__ Bh, const _Float16* __restrict__ Bl,
    float* __restrict__ C, int M, int K) {
    const int Nout = NT * 16;
    int lane = threadIdx.x & 31;
    int wid  = threadIdx.x >> 5;
    int mBase = blockIdx.x * 128 + wid * 16;
    if (mBase >= M) return;                   // wave-uniform: EXEC stays all-ones for WMMA
    int ln16 = lane & 15;
    int hiHalf = (lane >> 4) & 1;
    long rowA = mBase + ln16;
    if (rowA >= M) rowA = M - 1;              // clamp OOB loads; stores guarded below
    const _Float16* aHrow = Ah + rowA * (long)K;
    const _Float16* aLrow = Al + rowA * (long)K;
    int aoff = hiHalf * 16;                   // A: lanes>=16 hold K {16..31}
    int boff = hiHalf * 16;                   // B: lanes>=16 hold K {16..31}

    v8f acc[NT] = {};
    for (int k0 = 0; k0 < K; k0 += 32) {
        v8h ah0 = *(const v8h*)(aHrow + k0 + aoff);
        v8h ah1 = *(const v8h*)(aHrow + k0 + 8 + aoff);
        v8h al0 = *(const v8h*)(aLrow + k0 + aoff);
        v8h al1 = *(const v8h*)(aLrow + k0 + 8 + aoff);
        v16h a_hi = __builtin_shufflevector(ah0, ah1, 0,1,2,3,4,5,6,7,8,9,10,11,12,13,14,15);
        v16h a_lo = __builtin_shufflevector(al0, al1, 0,1,2,3,4,5,6,7,8,9,10,11,12,13,14,15);
#pragma unroll
        for (int nt = 0; nt < NT; ++nt) {
            long col = nt * 16 + ln16;
            v16h b_hi = *(const v16h*)(Bh + col * K + k0 + boff);
            v16h b_lo = *(const v16h*)(Bl + col * K + k0 + boff);
            acc[nt] = __builtin_amdgcn_wmma_f32_16x16x32_f16(false, a_hi, false, b_hi,
                                                             (short)0, acc[nt], false, false);
            acc[nt] = __builtin_amdgcn_wmma_f32_16x16x32_f16(false, a_hi, false, b_lo,
                                                             (short)0, acc[nt], false, false);
            acc[nt] = __builtin_amdgcn_wmma_f32_16x16x32_f16(false, a_lo, false, b_hi,
                                                             (short)0, acc[nt], false, false);
        }
    }
#pragma unroll
    for (int nt = 0; nt < NT; ++nt) {
#pragma unroll
        for (int r = 0; r < 8; ++r) {
            int row = mBase + r + hiHalf * 8;   // C layout: VGPR r -> row r (lanes<16) / r+8
            if (row < M) C[(long)row * Nout + nt * 16 + ln16] = acc[nt][r];
        }
    }
}

// ---------------- edge aggregation: out[dst] += norm * h[src], F = 128 ----------------

__global__ void aggregate_kernel(const float* __restrict__ h, const int* __restrict__ src,
                                 const int* __restrict__ dst, const float* __restrict__ nrm,
                                 float* __restrict__ out, int E) {
    int gw   = (int)(((long)blockIdx.x * blockDim.x + threadIdx.x) >> 5);
    int lane = threadIdx.x & 31;
    int nw   = (gridDim.x * blockDim.x) >> 5;
    for (int e = gw; e < E; e += nw) {
        int s = src[e], d = dst[e];
        float w = nrm[e];
        float4 v = ((const float4*)(h + (long)s * 128))[lane];
        float* o = out + (long)d * 128 + lane * 4;
        atomicAdd(o + 0, w * v.x);
        atomicAdd(o + 1, w * v.y);
        atomicAdd(o + 2, w * v.z);
        atomicAdd(o + 3, w * v.w);
    }
}

// ---------------- bias + ReLU (+ f16 split for next GEMM) ----------------

__global__ void bias_relu_split_kernel(const float* __restrict__ g, const float* __restrict__ bias,
                                       float* __restrict__ hf, _Float16* __restrict__ hh,
                                       _Float16* __restrict__ hl, long n, int F) {
    long i = (long)blockIdx.x * blockDim.x + threadIdx.x;
    if (i >= n) return;
    int f = (int)(i % F);
    float v = fmaxf(g[i] + bias[f], 0.0f);
    hf[i] = v;
    _Float16 hi = (_Float16)v;
    hh[i] = hi;
    hl[i] = (_Float16)(v - (float)hi);
}

__global__ void bias_relu_out_kernel(const float* __restrict__ g, const float* __restrict__ bias,
                                     float* __restrict__ out, long n, int F) {
    long i = (long)blockIdx.x * blockDim.x + threadIdx.x;
    if (i >= n) return;
    int f = (int)(i % F);
    out[i] = fmaxf(g[i] + bias[f], 0.0f);
}

// ---------------- host ----------------

static inline int gs(long n, int t) { return (int)((n + t - 1) / t); }

extern "C" void kernel_launch(void* const* d_in, const int* in_sizes, int n_in,
                              void* d_out, int out_size, void* d_ws, size_t ws_size,
                              hipStream_t stream) {
    (void)n_in; (void)out_size; (void)ws_size;

    const float* x    = (const float*)d_in[0];
    const int*   eidx = (const int*)d_in[1];
    const float* ew   = (const float*)d_in[2];
    const float* bng  = (const float*)d_in[3];
    const float* bnb  = (const float*)d_in[4];
    const float* W0   = (const float*)d_in[5];
    const float* b0   = (const float*)d_in[6];
    const float* W1   = (const float*)d_in[7];
    const float* b1   = (const float*)d_in[8];
    const float* W2   = (const float*)d_in[9];
    const float* b2   = (const float*)d_in[10];
    const float* pg   = (const float*)d_in[11];
    const float* pbeb = (const float*)d_in[12];
    const float* pW   = (const float*)d_in[13];
    const float* pb   = (const float*)d_in[14];

    const int N = in_sizes[0] / 256;   // 100000
    const int E = in_sizes[2];         // 600000
    const int* srcp = eidx;
    const int* dstp = eidx + E;

    char* ws = (char*)d_ws;
    size_t off = 0;
    auto alloc = [&](size_t bytes) -> char* {
        char* p = ws + off;
        off = (off + bytes + 255) & ~(size_t)255;
        return p;
    };

    float* deg   = (float*)alloc((size_t)N * 4);
    float* dis   = (float*)alloc((size_t)N * 4);
    float* nrm   = (float*)alloc((size_t)E * 4);
    float* sums  = (float*)alloc(512 * 4);
    float* scale = (float*)alloc(256 * 4);
    float* shift = (float*)alloc(256 * 4);
    _Float16* w0h = (_Float16*)alloc(256 * 128 * 2);
    _Float16* w0l = (_Float16*)alloc(256 * 128 * 2);
    _Float16* w1h = (_Float16*)alloc(128 * 128 * 2);
    _Float16* w1l = (_Float16*)alloc(128 * 128 * 2);
    _Float16* w2h = (_Float16*)alloc(128 * 128 * 2);
    _Float16* w2l = (_Float16*)alloc(128 * 128 * 2);
    _Float16* pwh = (_Float16*)alloc(128 * 64 * 2);
    _Float16* pwl = (_Float16*)alloc(128 * 64 * 2);
    _Float16* xh  = (_Float16*)alloc((size_t)N * 256 * 2);
    _Float16* xl  = (_Float16*)alloc((size_t)N * 256 * 2);
    float* g    = (float*)alloc((size_t)N * 128 * 4);
    float* agg  = (float*)alloc((size_t)N * 128 * 4);
    float* hf   = (float*)alloc((size_t)N * 128 * 4);
    _Float16* hh = (_Float16*)alloc((size_t)N * 128 * 2);
    _Float16* hl = (_Float16*)alloc((size_t)N * 128 * 2);

    const int T = 256;
    const long NF = (long)N * 128;

    // edge normalization: deg -> dis -> norm
    zero_kernel<<<gs(N, T), T, 0, stream>>>(deg, N);
    zero_kernel<<<1, T, 0, stream>>>(sums, 512);
    deg_kernel<<<gs(E, T), T, 0, stream>>>(ew, dstp, deg, E);
    dis_kernel<<<gs(N, T), T, 0, stream>>>(deg, dis, N);
    norm_kernel<<<gs(E, T), T, 0, stream>>>(srcp, dstp, ew, dis, nrm, E);

    // input batchnorm (256 features) + split to f16 hi/lo
    bn_stats_kernel<<<200, T, 0, stream>>>(x, sums, N, 256, 500);
    bn_finalize_kernel<<<1, T, 0, stream>>>(sums, bng, bnb, scale, shift, 256, 1.0f / N);
    bn_apply_split_kernel<<<gs((long)N * 256, T), T, 0, stream>>>(x, scale, shift, xh, xl,
                                                                  (long)N * 256, 256);

    // weight prep
    transpose_split_kernel<<<gs(256 * 128, T), T, 0, stream>>>(W0, w0h, w0l, 256, 128);
    transpose_split_kernel<<<gs(128 * 128, T), T, 0, stream>>>(W1, w1h, w1l, 128, 128);
    transpose_split_kernel<<<gs(128 * 128, T), T, 0, stream>>>(W2, w2h, w2l, 128, 128);
    transpose_split_kernel<<<gs(128 * 64, T), T, 0, stream>>>(pW, pwh, pwl, 128, 64);

    const int aggBlocks = gs((long)E * 32, T);

    // conv 1: 256 -> 128
    gemm_wmma_split<8><<<gs(N, 128), T, 0, stream>>>(xh, xl, w0h, w0l, g, N, 256);
    zero_kernel<<<gs(NF, T), T, 0, stream>>>(agg, NF);
    aggregate_kernel<<<aggBlocks, T, 0, stream>>>(g, srcp, dstp, nrm, agg, E);
    bias_relu_split_kernel<<<gs(NF, T), T, 0, stream>>>(agg, b0, hf, hh, hl, NF, 128);

    // conv 2: 128 -> 128
    gemm_wmma_split<8><<<gs(N, 128), T, 0, stream>>>(hh, hl, w1h, w1l, g, N, 128);
    zero_kernel<<<gs(NF, T), T, 0, stream>>>(agg, NF);
    aggregate_kernel<<<aggBlocks, T, 0, stream>>>(g, srcp, dstp, nrm, agg, E);
    bias_relu_split_kernel<<<gs(NF, T), T, 0, stream>>>(agg, b1, hf, hh, hl, NF, 128);

    // conv 3: 128 -> 128
    gemm_wmma_split<8><<<gs(N, 128), T, 0, stream>>>(hh, hl, w2h, w2l, g, N, 128);
    zero_kernel<<<gs(NF, T), T, 0, stream>>>(agg, NF);
    aggregate_kernel<<<aggBlocks, T, 0, stream>>>(g, srcp, dstp, nrm, agg, E);
    bias_relu_split_kernel<<<gs(NF, T), T, 0, stream>>>(agg, b2, hf, hh, hl, NF, 128);

    // projection head: BN(128) -> Linear(128->64) -> ReLU
    zero_kernel<<<1, T, 0, stream>>>(sums, 256);
    bn_stats_kernel<<<200, T, 0, stream>>>(hf, sums, N, 128, 500);
    bn_finalize_kernel<<<1, 128, 0, stream>>>(sums, pg, pbeb, scale, shift, 128, 1.0f / N);
    bn_apply_split_kernel<<<gs(NF, T), T, 0, stream>>>(hf, scale, shift, hh, hl, NF, 128);
    gemm_wmma_split<4><<<gs(N, 128), T, 0, stream>>>(hh, hl, pwh, pwl, g, N, 128);
    bias_relu_out_kernel<<<gs((long)N * 64, T), T, 0, stream>>>(g, pb, (float*)d_out,
                                                                (long)N * 64, 64);
}
